// GLAAttention_56349970924148
// MI455X (gfx1250) — matrix operand order repaired and checked by reading, not compiled, not run on hardware
//
#include <hip/hip_runtime.h>
#include <hip/hip_bf16.h>

typedef __attribute__((ext_vector_type(16))) __bf16 v16bf;
typedef __attribute__((ext_vector_type(8)))  float  v8f;

namespace gla {
constexpr int H   = 768;
constexpr int S   = 2048;
constexpr int B   = 4;
constexpr int C   = 64;
constexpr int NB  = S / C;   // 32 chunks
constexpr int NT  = B * S;   // 8192 token rows
constexpr int TV  = 32;      // v-columns owned per scan block
constexpr int LSF = 772;     // padded fp32 state stride (bank-conflict avoidance)
constexpr int LSB = 776;     // padded bf16 state stride
constexpr int LAT = 72;      // padded bf16 attn stride
}

union FragBF { v16bf v; int4 q[2]; };

// A-fragment (16x32 bf16, M=lane&15, halves hold K 0..7/16..23 | 8..15/24..31)
// from row-major storage: two 16B loads.
__device__ __forceinline__ v16bf load_rowA(const __bf16* p, int ld, int row, int k0, int half) {
  FragBF f;
  const __bf16* r = p + (size_t)row * ld + k0 + half * 8;
  f.q[0] = *(const int4*)(r);
  f.q[1] = *(const int4*)(r + 16);
  return f.v;
}
// B-fragment (32x16 bf16, N=lane&15, halves hold K 0..7/16..23 | 8..15/24..31)
// from column-major storage (column contiguous in K): two 16B loads.
__device__ __forceinline__ v16bf load_colB(const __bf16* p, int ld, int col, int k0, int half) {
  FragBF f;
  const __bf16* c = p + (size_t)col * ld + k0 + half * 8;
  f.q[0] = *(const int4*)(c);
  f.q[1] = *(const int4*)(c + 16);
  return f.v;
}
__device__ __forceinline__ v8f wmma_bf16(v16bf a, v16bf b, v8f c) {
  return __builtin_amdgcn_wmma_f32_16x16x32_bf16(false, a, false, b, (short)0, c, false, false);
}

// ---------------- prep: bf16 conversion + weight transpose ----------------
__global__ void k_prep(const float* __restrict__ x,
                       const float* __restrict__ Wq, const float* __restrict__ Wk,
                       const float* __restrict__ Wv,
                       __bf16* __restrict__ xb,
                       __bf16* __restrict__ WqT, __bf16* __restrict__ WkT,
                       __bf16* __restrict__ WvT) {
  using namespace gla;
  const int tid = blockIdx.x * blockDim.x + threadIdx.x;
  const int stride = gridDim.x * blockDim.x;
  for (int i = tid; i < NT * H; i += stride) xb[i] = (__bf16)x[i];
  for (int i = tid; i < H * H; i += stride) {
    const int n = i / H, k = i % H;          // WT[n][k] = W[k][n]
    WqT[i] = (__bf16)Wq[(size_t)k * H + n];
    WkT[i] = (__bf16)Wk[(size_t)k * H + n];
    WvT[i] = (__bf16)Wv[(size_t)k * H + n];
  }
}

// ---------------- QKV projection GEMM (WMMA bf16) ----------------
// grid = (NT/128, H/128, 3); 8 waves/block; each wave computes a 32x64 output
// tile (2 A-frags + 4 B-frags -> 8 WMMA per k-step; 1.5 b128 loads per WMMA).
// z=0 -> Q row-major; z=1 -> K row-major + K^T [B,H,S]; z=2 -> V^T [B,H,S].
__global__ void __launch_bounds__(256) k_qkv(
    const __bf16* __restrict__ xb,
    const __bf16* __restrict__ WqT, const __bf16* __restrict__ WkT,
    const __bf16* __restrict__ WvT,
    __bf16* __restrict__ Q, __bf16* __restrict__ K,
    __bf16* __restrict__ Kt, __bf16* __restrict__ Vt) {
  using namespace gla;
  const int lane = threadIdx.x & 31;
  const int wave = threadIdx.x >> 5;
  const int half = lane >> 4;
  const int l15  = lane & 15;
  const int rb   = blockIdx.x * 128;
  const int cbs  = blockIdx.y * 128;
  const int z    = blockIdx.z;
  const __bf16* WT = (z == 0) ? WqT : (z == 1) ? WkT : WvT;

  const int m0 = rb + (wave & 3) * 32;    // 2 m-tiles: m0, m0+16
  const int n0 = cbs + (wave >> 2) * 64;  // 4 n-tiles: n0 .. n0+48

  v8f acc[2][4];
#pragma unroll
  for (int i = 0; i < 2; ++i)
#pragma unroll
    for (int j = 0; j < 4; ++j) acc[i][j] = (v8f){0.f,0.f,0.f,0.f,0.f,0.f,0.f,0.f};

  for (int kk = 0; kk < H; kk += 32) {
    v16bf a0 = load_rowA(xb, H, m0 + l15,      kk, half);
    v16bf a1 = load_rowA(xb, H, m0 + 16 + l15, kk, half);
    v16bf b0 = load_colB(WT, H, n0 + l15,      kk, half);
    v16bf b1 = load_colB(WT, H, n0 + 16 + l15, kk, half);
    v16bf b2 = load_colB(WT, H, n0 + 32 + l15, kk, half);
    v16bf b3 = load_colB(WT, H, n0 + 48 + l15, kk, half);
    acc[0][0] = wmma_bf16(a0, b0, acc[0][0]);
    acc[0][1] = wmma_bf16(a0, b1, acc[0][1]);
    acc[0][2] = wmma_bf16(a0, b2, acc[0][2]);
    acc[0][3] = wmma_bf16(a0, b3, acc[0][3]);
    acc[1][0] = wmma_bf16(a1, b0, acc[1][0]);
    acc[1][1] = wmma_bf16(a1, b1, acc[1][1]);
    acc[1][2] = wmma_bf16(a1, b2, acc[1][2]);
    acc[1][3] = wmma_bf16(a1, b3, acc[1][3]);
  }

#pragma unroll
  for (int i = 0; i < 2; ++i) {
    const int sr0 = m0 + i * 16 + half * 8;      // 8 consecutive token rows
#pragma unroll
    for (int j = 0; j < 4; ++j) {
      const int col = n0 + j * 16 + l15;
      if (z == 0) {
#pragma unroll
        for (int r = 0; r < 8; ++r)
          Q[(size_t)(sr0 + r) * H + col] = (__bf16)acc[i][j][r];
      } else {
        // D-fragment rows are consecutive -> packed 16B transposed store.
        int4 p; __bf16* pb = (__bf16*)&p;
#pragma unroll
        for (int r = 0; r < 8; ++r) pb[r] = (__bf16)acc[i][j][r];
        const int bt = sr0 / S;
        const int sl = sr0 % S;
        __bf16* T = (z == 1) ? Kt : Vt;
        *(int4*)(T + ((size_t)bt * H + col) * S + sl) = p;
        if (z == 1) {
#pragma unroll
          for (int r = 0; r < 8; ++r)
            K[(size_t)(sr0 + r) * H + col] = (__bf16)acc[i][j][r];
        }
      }
    }
  }
}

// ---------------- gating: hidden = x @ W1 (fp32) ----------------
__global__ void k_hidden(const float* __restrict__ x, const float* __restrict__ W1,
                         float* __restrict__ H16) {
  using namespace gla;
  const int row = blockIdx.x * blockDim.x + threadIdx.x;
  if (row >= NT) return;
  float h[16];
#pragma unroll
  for (int j = 0; j < 16; ++j) h[j] = 0.f;
  const float* xr = x + (size_t)row * H;
  for (int d = 0; d < H; ++d) {
    const float xv = xr[d];
    const float* w = W1 + (size_t)d * 16;
#pragma unroll
    for (int j = 0; j < 16; ++j) h[j] += xv * w[j];
  }
#pragma unroll
  for (int j = 0; j < 16; ++j) H16[(size_t)row * 16 + j] = h[j];
}

// ---------------- gating: a[b,h] = mean_s sigmoid(h@W2 + b)^(1/16) ----------------
__global__ void k_alpha(const float* __restrict__ H16, const float* __restrict__ W2,
                        const float* __restrict__ bias, float* __restrict__ a) {
  using namespace gla;
  const int t = blockIdx.x * blockDim.x + threadIdx.x;
  if (t >= B * H) return;
  const int bb = t / H, col = t % H;
  float w2c[16];
#pragma unroll
  for (int j = 0; j < 16; ++j) w2c[j] = W2[(size_t)j * H + col];
  const float bc = bias[col];
  float acc = 0.f;
  const float* Hb = H16 + (size_t)bb * S * 16;
  for (int s = 0; s < S; ++s) {
    float dot = bc;
#pragma unroll
    for (int j = 0; j < 16; ++j) dot += Hb[(size_t)s * 16 + j] * w2c[j];
    const float sig = 1.0f / (1.0f + expf(-dot));
    acc += powf(sig, 0.0625f);
  }
  a[t] = acc * (1.0f / (float)S);
}

// ---------------- chunked scan: each block owns (batch, 32 v-columns) ----------------
__global__ void __launch_bounds__(256) k_scan(
    const __bf16* __restrict__ Q, const __bf16* __restrict__ K,
    const __bf16* __restrict__ Kt, const __bf16* __restrict__ Vt,
    const float* __restrict__ a, float* __restrict__ out) {
  using namespace gla;
  extern __shared__ char smem[];
  float*  Sf = (float*)smem;                                          // [TV][LSF] fp32 state, col-major
  __bf16* Sb = (__bf16*)(smem + (size_t)TV * LSF * 4);                // [TV][LSB] bf16 shadow
  __bf16* At = (__bf16*)(smem + (size_t)TV * LSF * 4 + (size_t)TV * LSB * 2); // [C][LAT] attn bf16

  const int lane = threadIdx.x & 31;
  const int wave = threadIdx.x >> 5;
  const int half = lane >> 4;
  const int l15  = lane & 15;
  const int bb   = blockIdx.y;
  const int vb   = blockIdx.x * TV;

  for (int i = threadIdx.x; i < TV * LSF; i += 256) Sf[i] = 0.0f;

  const float av0 = a[bb * H + vb + l15];        // decay for n-tile 0 (per column)
  const float av1 = a[bb * H + vb + 16 + l15];   // decay for n-tile 1
  const float scale = 0.03608439182435161f;      // 1/sqrt(768)

  const __bf16* Qp  = Q  + (size_t)bb * S * H;
  const __bf16* Kp  = K  + (size_t)bb * S * H;
  const __bf16* Ktp = Kt + (size_t)bb * H * S;
  const __bf16* Vtp = Vt + (size_t)bb * H * S;

  __syncthreads();

  for (int ch = 0; ch < NB; ++ch) {
    const int cb = ch * C;

    // ---- phase 1: attn = tril(q k^T)*scale -> At (bf16) ----
    // A (Q) fragment shared across both e-tiles of this wave.
    {
      const int mt = wave & 3;
      const int arow = cb + mt * 16 + l15;
      const int e0 = (wave >> 2) * 32;
      v8f ac0 = {0.f,0.f,0.f,0.f,0.f,0.f,0.f,0.f};
      v8f ac1 = {0.f,0.f,0.f,0.f,0.f,0.f,0.f,0.f};
      for (int kk = 0; kk < H; kk += 32) {
        v16bf af  = load_rowA(Qp, H, arow, kk, half);
        v16bf bf0 = load_colB(Kp, H, cb + e0 + l15,      kk, half);
        v16bf bf1 = load_colB(Kp, H, cb + e0 + 16 + l15, kk, half);
        ac0 = wmma_bf16(af, bf0, ac0);
        ac1 = wmma_bf16(af, bf1, ac1);
      }
      const int cr = mt * 16 + half * 8;
#pragma unroll
      for (int r = 0; r < 8; ++r) {
        const int c = cr + r;
        const int ea = e0 + l15;
        const int eb = e0 + 16 + l15;
        At[c * LAT + ea] = (__bf16)((ea <= c) ? ac0[r] * scale : 0.0f);
        At[c * LAT + eb] = (__bf16)((eb <= c) ? ac1[r] * scale : 0.0f);
      }
    }

    // V^T B-fragments for this block's v columns (reused in phases 2 and 3)
    v16bf bv[2][2];
#pragma unroll
    for (int n = 0; n < 2; ++n)
#pragma unroll
      for (int kid = 0; kid < 2; ++kid)
        bv[n][kid] = load_colB(Vtp, S, vb + n * 16 + l15, cb + kid * 32, half);

    // ---- phase 2: S = S*a + K^T V (fp32 in LDS), refresh bf16 shadow ----
    for (int mi = 0; mi < 6; ++mi) {
      const int mt = wave * 6 + mi;                       // d-tile 0..47
      v16bf ak0 = load_rowA(Ktp, S, mt * 16 + l15, cb,      half);
      v16bf ak1 = load_rowA(Ktp, S, mt * 16 + l15, cb + 32, half);
      if (ch + 1 < NB) {
        // global_prefetch: pull next chunk's K^T rows toward the caches
        __builtin_prefetch(Ktp + (size_t)(mt * 16 + l15) * S + cb + C, 0, 1);
      }
#pragma unroll
      for (int n = 0; n < 2; ++n) {
        const int vl = n * 16 + l15;
        float* sp = Sf + (size_t)vl * LSF + mt * 16 + half * 8; // 8 consecutive d
        float4 c0 = *(float4*)sp;
        float4 c1 = *(float4*)(sp + 4);
        const float av = n ? av1 : av0;
        v8f cc;
        cc[0]=c0.x*av; cc[1]=c0.y*av; cc[2]=c0.z*av; cc[3]=c0.w*av;
        cc[4]=c1.x*av; cc[5]=c1.y*av; cc[6]=c1.z*av; cc[7]=c1.w*av;
        cc = wmma_bf16(ak0, bv[n][0], cc);
        cc = wmma_bf16(ak1, bv[n][1], cc);
        *(float4*)sp       = make_float4(cc[0], cc[1], cc[2], cc[3]);
        *(float4*)(sp + 4) = make_float4(cc[4], cc[5], cc[6], cc[7]);
        int4 pk; __bf16* pb = (__bf16*)&pk;
#pragma unroll
        for (int r = 0; r < 8; ++r) pb[r] = (__bf16)cc[r];
        *(int4*)(Sb + (size_t)vl * LSB + mt * 16 + half * 8) = pk;
      }
    }

    __syncthreads();

    // ---- phase 3: out = q @ S + attn @ v ----
    {
      const int mt = wave & 3;
      const int n  = wave >> 2;
      const int vl = n * 16 + l15;
      v8f acc = {0.f,0.f,0.f,0.f,0.f,0.f,0.f,0.f};
      const int qrow = cb + mt * 16 + l15;
      for (int kk = 0; kk < H; kk += 32) {
        v16bf af = load_rowA(Qp, H, qrow, kk, half);
        v16bf bf = load_colB(Sb, LSB, vl, kk, half);     // LDS bf16 shadow state
        acc = wmma_bf16(af, bf, acc);
      }
#pragma unroll
      for (int kid = 0; kid < 2; ++kid) {
        v16bf af = load_rowA(At, LAT, mt * 16 + l15, kid * 32, half); // LDS attn
        acc = wmma_bf16(af, bv[n][kid], acc);
      }
      const int s0  = cb + mt * 16 + half * 8;
      const int col = vb + vl;
      float* op = out + ((size_t)bb * S + s0) * H + col;
#pragma unroll
      for (int r = 0; r < 8; ++r) op[(size_t)r * H] = acc[r];
    }
    __syncthreads();
  }
}

// ---------------- host ----------------
extern "C" void kernel_launch(void* const* d_in, const int* in_sizes, int n_in,
                              void* d_out, int out_size, void* d_ws, size_t ws_size,
                              hipStream_t stream) {
  using namespace gla;
  (void)in_sizes; (void)n_in; (void)out_size; (void)ws_size;
  const float* x    = (const float*)d_in[0];
  const float* Wq   = (const float*)d_in[1];
  const float* Wk   = (const float*)d_in[2];
  const float* Wv   = (const float*)d_in[3];
  const float* W1   = (const float*)d_in[4];
  const float* W2   = (const float*)d_in[5];
  const float* bias = (const float*)d_in[6];
  float* out = (float*)d_out;

  char* w = (char*)d_ws;
  size_t o = 0;
  auto take = [&](size_t n) { char* p = w + o; o += (n + 255) & ~(size_t)255; return p; };
  __bf16* xb  = (__bf16*)take((size_t)NT * H * 2);
  __bf16* WqT = (__bf16*)take((size_t)H * H * 2);
  __bf16* WkT = (__bf16*)take((size_t)H * H * 2);
  __bf16* WvT = (__bf16*)take((size_t)H * H * 2);
  __bf16* Qb  = (__bf16*)take((size_t)NT * H * 2);
  __bf16* Kb  = (__bf16*)take((size_t)NT * H * 2);
  __bf16* Kt  = (__bf16*)take((size_t)NT * H * 2);
  __bf16* Vt  = (__bf16*)take((size_t)NT * H * 2);
  float*  H16 = (float*)take((size_t)NT * 16 * 4);
  float*  ad  = (float*)take((size_t)B * H * 4);

  k_prep<<<2048, 256, 0, stream>>>(x, Wq, Wk, Wv, xb, WqT, WkT, WvT);
  k_qkv<<<dim3(NT / 128, H / 128, 3), 256, 0, stream>>>(xb, WqT, WkT, WvT, Qb, Kb, Kt, Vt);
  k_hidden<<<NT / 256, 256, 0, stream>>>(x, W1, H16);
  k_alpha<<<(B * H) / 256, 256, 0, stream>>>(H16, W2, bias, ad);

  const size_t smem = (size_t)TV * LSF * 4 + (size_t)TV * LSB * 2 + (size_t)C * LAT * 2; // ~154 KB
  k_scan<<<dim3(H / TV, B), 256, smem, stream>>>(Qb, Kb, Kt, Vt, ad, out);
}
